// EMGTransformer_42039139893362
// MI455X (gfx1250) — matrix operand order repaired and checked by reading, not compiled, not run on hardware
//
#include <hip/hip_runtime.h>

// ---------------------------------------------------------------------------
// Types
// ---------------------------------------------------------------------------
typedef __bf16 v16bf __attribute__((ext_vector_type(16)));
typedef float  v8f   __attribute__((ext_vector_type(8)));

#define TM 128
#define TN 64
#define TK 32
#define LSTR 40   // LDS row stride in halves (80 B, multiple of 16 B)
#define NTHR 256

#if defined(__has_builtin)
#if __has_builtin(__builtin_amdgcn_global_load_async_to_lds_b128) && \
    __has_builtin(__builtin_amdgcn_s_wait_asynccnt)
#define USE_ASYNC 1
#endif
#endif

#ifdef USE_ASYNC
typedef int v4i_ __attribute__((__vector_size__(4 * sizeof(int))));
typedef __attribute__((address_space(1))) v4i_* g4p_t;   // global int4*
typedef __attribute__((address_space(3))) v4i_* l4p_t;   // LDS int4*
#endif

struct ZOfs {            // batched base offsets: ofs = (z/dv)*outer + (z%dv)*inner
  long long aO, aI, bO, bI, cO, cI;
  int dv;
};

struct Epi {
  const float* bias;
  const float* bnG; const float* bnB; const float* bnM; const float* bnV;
  const float* add;          // residual, f32, same ldc / batch offsets as C
  float*  outF;              // optional f32 output
  __bf16* outH;              // optional bf16 output
  int gelu;
};

// ---------------------------------------------------------------------------
// WMMA GEMM:  C[M,N] = A[M,K](bf16) x B[N,K]^T  (+epilogue)
//   B is [N][K] row-major (torch linear weights / K matrices / V^T)
//   BF32 : B elements are f32 (converted to bf16 while staging) else bf16
// Tile: 128x64, BK=32, 256 threads (8 waves); wave w does rows [16w,16w+16).
// Interior tiles stage via GLOBAL_LOAD_ASYNC_TO_LDS_B128 (bf16 operands) or
// 16-B vector loads (+cvt for f32 weights); edges use a scalar fallback.
// ---------------------------------------------------------------------------
template <int BF32>
__global__ __launch_bounds__(NTHR) void k_gemm(
    const __bf16* __restrict__ A, const void* __restrict__ Bp,
    int M, int N, int K, int lda, int ldb, int ldc, ZOfs zo, Epi ep)
{
  __shared__ __align__(16) __bf16 As[TM * LSTR];
  __shared__ __align__(16) __bf16 Bs[TN * LSTR];

  const int tid = threadIdx.x;
  const int z   = blockIdx.z;
  const long long zq = z / zo.dv, zr = z % zo.dv;
  A += zq * zo.aO + zr * zo.aI;
  const long long bofs = zq * zo.bO + zr * zo.bI;
  const long long cofs = zq * zo.cO + zr * zo.cI;
  const float*  Bf = (const float*)Bp + (BF32 ? bofs : 0);
  const __bf16* Bh = (const __bf16*)Bp + (BF32 ? 0 : bofs);

  const int mBase = blockIdx.y * TM;
  const int nBase = blockIdx.x * TN;

  const int wave = tid >> 5, lane = tid & 31;
  const int lm = lane & 15, kh = lane >> 4;

  v8f zero = {0, 0, 0, 0, 0, 0, 0, 0};
  v8f acc[4];
#pragma unroll
  for (int t = 0; t < 4; ++t) acc[t] = zero;

  const int rA = tid >> 1;  const int sA = tid & 1;   // A: 128 rows x 2 halves-of-row
  const int rB = tid >> 2;  const int sB = tid & 3;   // B: 64 rows x 4 segments of 8

  const bool interior =
      (mBase + TM <= M) && (nBase + TN <= N) &&
      ((lda & 7) == 0) && ((((size_t)A) & 15) == 0) &&
      (BF32 ? (((ldb & 3) == 0) && ((((size_t)Bf) & 15) == 0))
            : (((ldb & 7) == 0) && ((((size_t)Bh) & 15) == 0)));

  const __bf16* aRow = A + (long long)(mBase + rA) * lda + sA * 16;

  for (int k0 = 0; k0 < K; k0 += TK) {
    if (interior && (k0 + TK <= K)) {
      // ---- fast staging ----
      {
        const __bf16* gsrc = aRow + k0;
        __bf16* ldst = &As[rA * LSTR + sA * 16];
#ifdef USE_ASYNC
        __builtin_amdgcn_global_load_async_to_lds_b128((g4p_t)gsrc, (l4p_t)ldst, 0, 0);
        __builtin_amdgcn_global_load_async_to_lds_b128((g4p_t)(gsrc + 8), (l4p_t)(ldst + 8), 0, 0);
#else
        const uint4* src = (const uint4*)gsrc;
        uint4 a0 = src[0], a1 = src[1];
        uint4* dst = (uint4*)ldst;
        dst[0] = a0; dst[1] = a1;
#endif
        if (k0 + TK < K) __builtin_prefetch(gsrc + TK, 0, 1);
      }
      if (BF32) {
        const float4* src = (const float4*)(Bf + (long long)(nBase + rB) * ldb + k0 + sB * 8);
        float4 f0 = src[0], f1 = src[1];
        union { __bf16 h[8]; uint4 u; } pk;
        pk.h[0] = (__bf16)f0.x; pk.h[1] = (__bf16)f0.y; pk.h[2] = (__bf16)f0.z; pk.h[3] = (__bf16)f0.w;
        pk.h[4] = (__bf16)f1.x; pk.h[5] = (__bf16)f1.y; pk.h[6] = (__bf16)f1.z; pk.h[7] = (__bf16)f1.w;
        *(uint4*)(&Bs[rB * LSTR + sB * 8]) = pk.u;
      } else {
        const __bf16* gsrc = Bh + (long long)(nBase + rB) * ldb + k0 + sB * 8;
        __bf16* ldst = &Bs[rB * LSTR + sB * 8];
#ifdef USE_ASYNC
        __builtin_amdgcn_global_load_async_to_lds_b128((g4p_t)gsrc, (l4p_t)ldst, 0, 0);
#else
        *(uint4*)ldst = *(const uint4*)gsrc;
#endif
      }
    } else {
      // ---- scalar fallback (tile edges, K tails) ----
      {
        int gm = mBase + rA;
#pragma unroll
        for (int i = 0; i < 16; ++i) {
          int kl = sA * 16 + i, kg = k0 + kl;
          __bf16 v = (__bf16)0.0f;
          if (gm < M && kg < K) v = A[(long long)gm * lda + kg];
          As[rA * LSTR + kl] = v;
        }
      }
      {
        int gn = nBase + rB;
#pragma unroll
        for (int i = 0; i < 8; ++i) {
          int kl = sB * 8 + i, kg = k0 + kl;
          __bf16 v = (__bf16)0.0f;
          if (gn < N && kg < K) {
            long long idx = (long long)gn * ldb + kg;
            v = BF32 ? (__bf16)Bf[idx] : Bh[idx];
          }
          Bs[rB * LSTR + kl] = v;
        }
      }
    }
#ifdef USE_ASYNC
    __builtin_amdgcn_s_wait_asynccnt(0);
#endif
    __syncthreads();

    // ---- WMMA: one A frag (16x32), preload 4 B frags, 4 back-to-back WMMAs ----
    union { v16bf v; uint4 u[2]; } af;
    const __bf16* arow = &As[(wave * 16 + lm) * LSTR];
    af.u[0] = *(const uint4*)(arow + kh * 8);
    af.u[1] = *(const uint4*)(arow + 16 + kh * 8);
    union { v16bf v; uint4 u[2]; } bfg[4];
#pragma unroll
    for (int nt = 0; nt < 4; ++nt) {
      const __bf16* brow = &Bs[(nt * 16 + lm) * LSTR];
      bfg[nt].u[0] = *(const uint4*)(brow + kh * 8);
      bfg[nt].u[1] = *(const uint4*)(brow + 16 + kh * 8);
    }
#pragma unroll
    for (int nt = 0; nt < 4; ++nt)
      acc[nt] = __builtin_amdgcn_wmma_f32_16x16x32_bf16(
          false, af.v, false, bfg[nt].v, (short)0, acc[nt], false, false);
    __syncthreads();
  }

  // ---- epilogue: C layout VGPR i -> row = wave*16 + kh*8 + i, col = lm ----
#pragma unroll
  for (int nt = 0; nt < 4; ++nt) {
#pragma unroll
    for (int i = 0; i < 8; ++i) {
      int gm = mBase + wave * 16 + kh * 8 + i;
      int gn = nBase + nt * 16 + lm;
      if (gm < M && gn < N) {
        float x = acc[nt][i];
        if (ep.bias) x += ep.bias[gn];
        if (ep.bnG)
          x = (x - ep.bnM[gn]) * (ep.bnG[gn] * rsqrtf(ep.bnV[gn] + 1e-5f)) + ep.bnB[gn];
        if (ep.add)  x += ep.add[cofs + (long long)gm * ldc + gn];
        if (ep.gelu) x = 0.5f * x * (1.0f + erff(x * 0.70710678118f));
        long long o = cofs + (long long)gm * ldc + gn;
        if (ep.outF) ep.outF[o] = x;
        if (ep.outH) ep.outH[o] = (__bf16)x;
      }
    }
  }
}

// ---------------------------------------------------------------------------
// f32 -> bf16 elementwise convert
// ---------------------------------------------------------------------------
__global__ void k_cvt(const float* __restrict__ s, __bf16* __restrict__ d, int n)
{
  int i = blockIdx.x * blockDim.x + threadIdx.x;
  if (i < n) d[i] = (__bf16)s[i];
}

// ---------------------------------------------------------------------------
// im2col for 1D conv, activations in [B][T][C] layout (bf16)
// ---------------------------------------------------------------------------
__global__ void k_im2col(const __bf16* __restrict__ src, __bf16* __restrict__ dst,
                         int Bn, int Tin, int Tout, int C, int K, int stride, int pad)
{
  long long idx = (long long)blockIdx.x * blockDim.x + threadIdx.x;
  long long total = (long long)Bn * Tout * C * K;
  if (idx >= total) return;
  int col = (int)(idx % (C * K));
  long long row = idx / (C * K);
  int b  = (int)(row / Tout);
  int to = (int)(row % Tout);
  int c = col / K, kk = col % K;
  int t = to * stride - pad + kk;
  __bf16 v = (__bf16)0.0f;
  if (t >= 0 && t < Tin) v = src[((long long)b * Tin + t) * C + c];
  dst[idx] = v;
}

// ---------------------------------------------------------------------------
// V^T materialization: VT[z=(b*6+h)][d][n] = V[b,h,n,d] from qkv rows
// ---------------------------------------------------------------------------
__global__ void k_transposeV(const __bf16* __restrict__ qkv, __bf16* __restrict__ VT)
{
  long long i = (long long)blockIdx.x * blockDim.x + threadIdx.x;
  if (i >= 48LL * 64 * 512) return;
  int n = (int)(i & 511);
  int d = (int)((i >> 9) & 63);
  int zz = (int)(i >> 15);
  int b = zz / 6, h = zz % 6;
  VT[i] = qkv[((long long)(b * 512 + n)) * 1152 + 768 + h * 64 + d];
}

// ---------------------------------------------------------------------------
// LayerNorm over E=384, rows = 4096
// ---------------------------------------------------------------------------
__global__ __launch_bounds__(128) void k_layernorm(
    const float* __restrict__ X, const float* __restrict__ g,
    const float* __restrict__ b, float* __restrict__ outF, __bf16* __restrict__ outH)
{
  const int E = 384;
  __shared__ float red[128];
  int row = blockIdx.x, tid = threadIdx.x;
  const float* x = X + (long long)row * E;
  float s = 0.f;
  for (int j = tid; j < E; j += 128) s += x[j];
  red[tid] = s; __syncthreads();
  for (int st = 64; st > 0; st >>= 1) { if (tid < st) red[tid] += red[tid + st]; __syncthreads(); }
  float mean = red[0] / E; __syncthreads();
  float v = 0.f;
  for (int j = tid; j < E; j += 128) { float d = x[j] - mean; v += d * d; }
  red[tid] = v; __syncthreads();
  for (int st = 64; st > 0; st >>= 1) { if (tid < st) red[tid] += red[tid + st]; __syncthreads(); }
  float inv = rsqrtf(red[0] / E + 1e-5f);
  for (int j = tid; j < E; j += 128) {
    float y = (x[j] - mean) * inv * g[j] + b[j];
    long long o = (long long)row * E + j;
    if (outF) outF[o] = y;
    if (outH) outH[o] = (__bf16)y;
  }
}

// ---------------------------------------------------------------------------
// softmax over keys with relative-position gather (faithful skew):
//   pos(q,k) = (k-q in [-99,99]) ? R[h][q*8+b][k-q+99] : -1e8
// ---------------------------------------------------------------------------
__global__ __launch_bounds__(128) void k_softmax_pos(
    const float* __restrict__ S, const float* __restrict__ R, __bf16* __restrict__ P)
{
  const int Nn = 512, Hh = 6, Bb = 8, RLD = 201;
  __shared__ float red[128];
  int row = blockIdx.x;
  int zz = row >> 9, q = row & 511;
  int b = zz / Hh, h = zz % Hh;
  const float* srow = S + (long long)row * Nn;
  const float* rrow = R + ((long long)h * 4096 + (q * Bb + b)) * RLD;
  int tid = threadIdx.x;
  float vals[4];
  float mx = -3.0e38f;
#pragma unroll
  for (int j = 0; j < 4; ++j) {
    int k = tid + j * 128;
    int rel = k - q;
    float p = (rel >= -99 && rel <= 99) ? rrow[rel + 99] : -1.0e8f;
    float v = srow[k] * 0.125f + p;
    vals[j] = v; mx = fmaxf(mx, v);
  }
  red[tid] = mx; __syncthreads();
  for (int st = 64; st > 0; st >>= 1) { if (tid < st) red[tid] = fmaxf(red[tid], red[tid + st]); __syncthreads(); }
  mx = red[0]; __syncthreads();
  float s = 0.f;
#pragma unroll
  for (int j = 0; j < 4; ++j) { vals[j] = __expf(vals[j] - mx); s += vals[j]; }
  red[tid] = s; __syncthreads();
  for (int st = 64; st > 0; st >>= 1) { if (tid < st) red[tid] += red[tid + st]; __syncthreads(); }
  float inv = 1.0f / red[0];
#pragma unroll
  for (int j = 0; j < 4; ++j) {
    int k = tid + j * 128;
    P[(long long)row * Nn + k] = (__bf16)(vals[j] * inv);
  }
}

// ---------------------------------------------------------------------------
// Host side
// ---------------------------------------------------------------------------
static inline dim3 gg(int M, int N, int Z) { return dim3((N + 63) / 64, (M + 127) / 128, Z); }

extern "C" void kernel_launch(void* const* d_in, const int* in_sizes, int n_in,
                              void* d_out, int out_size, void* d_ws, size_t ws_size,
                              hipStream_t stream)
{
  (void)in_sizes; (void)n_in; (void)out_size; (void)ws_size;
  auto W = [&](int i) { return (const float*)d_in[i]; };
  char* ws = (char*)d_ws;
  const size_t MB = 1ull << 20;

  // workspace layout (regions reused across phases)
  __bf16* BIGh = (__bf16*)(ws + 0);          // im2col (<=38MB)
  float*  Sf   = (float*) (ws + 0);          // scores (50.4MB) / Y (6.3MB)
  __bf16* R1h  = (__bf16*)(ws + 51 * MB);    // conv r1 (12.6MB)
  __bf16* Pbh  = (__bf16*)(ws + 51 * MB);    // attention probs bf16 (25.2MB)
  __bf16* ACTh = (__bf16*)(ws + 77 * MB);    // conv ping (12.6MB)
  float*  Rf   = (float*) (ws + 90 * MB);    // rel-pos R (19.8MB)
  float*  RESf = (float*) (ws + 110 * MB);   // r2 / residual f32 (25.2MB)
  __bf16* QKV  = (__bf16*)(ws + 136 * MB);   // qkv bf16 (9.5MB)
  __bf16* XH   = (__bf16*)(ws + 146 * MB);   // x bf16 (3.1MB)
  __bf16* Oh   = (__bf16*)(ws + 150 * MB);   // attn out bf16 (3.1MB)
  __bf16* Hh   = (__bf16*)(ws + 154 * MB);   // ffn hidden bf16 (12.6MB)
  __bf16* VTh  = (__bf16*)(ws + 167 * MB);   // V^T bf16 (3.1MB)

  ZOfs z0{0, 0, 0, 0, 0, 0, 1};

  // ---- convert x_raw (B,4096,8) f32 -> bf16 at XH ----
  {
    int n = 8 * 4096 * 8;
    k_cvt<<<(n + 255) / 256, 256, 0, stream>>>(W(1), XH, n);
  }

  // ---- conv resblocks (post-activation, stride 2) ----
  const __bf16* cur = XH;
  int Tin = 4096;
  for (int blk = 0; blk < 3; ++blk) {
    int ci = (blk == 0) ? 8 : 384;
    int Tout = Tin / 2;
    int bi = 3 + blk * 18;      // w1,b1,bn1{g,b,m,v},w2,b2,bn2{...},ws,bs,bns{...}
    int M = 8 * Tout;
    long long tot;

    // r1 = gelu(bn1(conv1(x)))
    int Kd = ci * 3;
    tot = (long long)M * Kd;
    k_im2col<<<(unsigned)((tot + 255) / 256), 256, 0, stream>>>(cur, BIGh, 8, Tin, Tout, ci, 3, 2, 1);
    Epi e1{}; e1.bias = W(bi + 1); e1.bnG = W(bi + 2); e1.bnB = W(bi + 3);
    e1.bnM = W(bi + 4); e1.bnV = W(bi + 5); e1.gelu = 1; e1.outH = R1h;
    k_gemm<1><<<gg(M, 384, 1), NTHR, 0, stream>>>(BIGh, W(bi), M, 384, Kd, Kd, Kd, 384, z0, e1);

    // r2 = bn2(conv2(r1))   (f32)
    tot = (long long)M * 1152;
    k_im2col<<<(unsigned)((tot + 255) / 256), 256, 0, stream>>>(R1h, BIGh, 8, Tout, Tout, 384, 3, 1, 1);
    Epi e2{}; e2.bias = W(bi + 7); e2.bnG = W(bi + 8); e2.bnB = W(bi + 9);
    e2.bnM = W(bi + 10); e2.bnV = W(bi + 11); e2.outF = RESf;
    k_gemm<1><<<gg(M, 384, 1), NTHR, 0, stream>>>(BIGh, W(bi + 6), M, 384, 1152, 1152, 1152, 384, z0, e2);

    // out = gelu(bns(convs(x)) + r2)
    tot = (long long)M * ci;
    k_im2col<<<(unsigned)((tot + 255) / 256), 256, 0, stream>>>(cur, BIGh, 8, Tin, Tout, ci, 1, 2, 0);
    Epi e3{}; e3.bias = W(bi + 13); e3.bnG = W(bi + 14); e3.bnB = W(bi + 15);
    e3.bnM = W(bi + 16); e3.bnV = W(bi + 17); e3.add = RESf; e3.gelu = 1; e3.outH = ACTh;
    k_gemm<1><<<gg(M, 384, 1), NTHR, 0, stream>>>(BIGh, W(bi + 12), M, 384, ci, ci, ci, 384, z0, e3);

    cur = ACTh; Tin = Tout;
  }

  // ---- x = conv_out @ w_raw_in^T + b  -> RESf (f32 residual), XH (bf16) ----
  {
    Epi er{}; er.bias = W(58); er.outF = RESf; er.outH = XH;
    k_gemm<1><<<gg(4096, 384, 1), NTHR, 0, stream>>>(cur, W(57), 4096, 384, 384, 384, 384, 384, z0, er);
  }

  // ---- transformer layers ----
  for (int L = 0; L < 8; ++L) {
    int bi = 59 + L * 13;  // qkv_w,qkv_b,proj_w,proj_b,rel_emb,ln1_g,ln1_b,ln2_g,ln2_b,fc1_w,fc1_b,fc2_w,fc2_b

    // qkv
    Epi eq{}; eq.bias = W(bi + 1); eq.outH = QKV;
    k_gemm<1><<<gg(4096, 1152, 1), NTHR, 0, stream>>>(XH, W(bi), 4096, 1152, 384, 384, 384, 1152, z0, eq);

    // R[h][idx][rel] = q4 . emb   (batched over h)
    ZOfs zr{0, 64, 0, 201 * 64, 0, 4096LL * 201, 6};
    Epi eR{}; eR.outF = Rf;
    k_gemm<1><<<gg(4096, 201, 6), NTHR, 0, stream>>>(QKV, W(bi + 4), 4096, 201, 64, 1152, 64, 201, zr, eR);

    // S = Q . K^T  (batched over b*H+h)
    ZOfs zs{512LL * 1152, 64, 512LL * 1152, 64, 6LL * 512 * 512, 512LL * 512, 6};
    Epi eS{}; eS.outF = Sf;
    k_gemm<0><<<gg(512, 512, 48), NTHR, 0, stream>>>(QKV, (const void*)(QKV + 384),
                                                     512, 512, 64, 1152, 1152, 512, zs, eS);

    // P = softmax(S/8 + pos)
    k_softmax_pos<<<48 * 512, 128, 0, stream>>>(Sf, Rf, Pbh);

    // V^T then O = P . V  (both operands T-layout now)
    {
      long long n = 48LL * 64 * 512;
      k_transposeV<<<(unsigned)((n + 255) / 256), 256, 0, stream>>>(QKV, VTh);
    }
    ZOfs za{6LL * 512 * 512, 512LL * 512, 6LL * 64 * 512, 64LL * 512, 512LL * 384, 64, 6};
    Epi eO{}; eO.outH = Oh;
    k_gemm<0><<<gg(512, 64, 48), NTHR, 0, stream>>>(Pbh, (const void*)VTh,
                                                    512, 64, 512, 512, 512, 384, za, eO);

    // y1 = x + O @ proj^T + b   -> Sf (as Y)
    Epi eP{}; eP.bias = W(bi + 3); eP.add = RESf; eP.outF = Sf;
    k_gemm<1><<<gg(4096, 384, 1), NTHR, 0, stream>>>(Oh, W(bi + 2), 4096, 384, 384, 384, 384, 384, z0, eP);

    // x1 = LN1(y1)  -> RESf (f32) + XH (bf16)
    k_layernorm<<<4096, 128, 0, stream>>>(Sf, W(bi + 5), W(bi + 6), RESf, XH);

    // h = gelu(x1 @ fc1^T + b)
    Epi e4{}; e4.bias = W(bi + 10); e4.gelu = 1; e4.outH = Hh;
    k_gemm<1><<<gg(4096, 1536, 1), NTHR, 0, stream>>>(XH, W(bi + 9), 4096, 1536, 384, 384, 384, 1536, z0, e4);

    // y2 = x1 + h @ fc2^T + b  -> Sf
    Epi e5{}; e5.bias = W(bi + 12); e5.add = RESf; e5.outF = Sf;
    k_gemm<1><<<gg(4096, 384, 1), NTHR, 0, stream>>>(Hh, W(bi + 11), 4096, 384, 1536, 1536, 1536, 384, z0, e5);

    // x = LN2(y2)
    k_layernorm<<<4096, 128, 0, stream>>>(Sf, W(bi + 7), W(bi + 8), RESf, XH);
  }

  // ---- head: out = x @ w_out^T + b  (f32, ldc = 38) ----
  {
    Epi ef{}; ef.bias = W(164); ef.outF = (float*)d_out;
    k_gemm<1><<<gg(4096, 38, 1), NTHR, 0, stream>>>(XH, W(163), 4096, 38, 384, 384, 384, 38, z0, ef);
  }
}
